// FixedSwinBlock3D_13821204758705
// MI455X (gfx1250) — matrix operand order repaired and checked by reading, not compiled, not run on hardware
//
#include <hip/hip_runtime.h>

// ---------------- types ----------------
typedef __attribute__((ext_vector_type(16))) __bf16          v16bf;
typedef __attribute__((ext_vector_type(16))) unsigned short  v16us;
typedef __attribute__((ext_vector_type(8)))  float           v8f;
typedef __attribute__((ext_vector_type(4)))  float           v4f;
typedef __attribute__((ext_vector_type(4)))  unsigned int    v4u;
typedef __attribute__((ext_vector_type(8)))  int             v8i;
typedef __attribute__((ext_vector_type(4)))  int             v4i;

// ---------------- problem constants ----------------
#define P_TOK   110592      // 48^3 tokens
#define C_DIM   96
#define T_WIN   216         // 6^3 tokens per window
#define T_PAD   224         // 14 * 16
#define HEADS   3
#define HD      32
#define QKV_N   288
#define HID     384
#define SCALE_F 0.17677669529663687f   // 32^-0.5

// union region bytes in attention kernel: max(sXN+sWqkvT = 98304, sP+sAO = 100352)
#define REGION_BYTES 100352

// ---------------- helpers ----------------
// hardware f32 -> bf16 conversion (v_cvt_*bf16*), RTNE per CDNA5 VALU rules
static __device__ __forceinline__ unsigned short f2bf(float f) {
  __bf16 h = (__bf16)f;
  return __builtin_bit_cast(unsigned short, h);
}

// A fragment (16x32 bf16): lane L holds row M=L&15; K-halves interleave:
// lanes 0-15 -> K {0..7,16..23}, lanes 16-31 -> K {8..15,24..31}
static __device__ __forceinline__ v16bf load_a(const unsigned short* base, int lda, int lane) {
  const unsigned short* r = base + (lane & 15) * lda + (lane >> 4) * 8;
  v16us t;
  *reinterpret_cast<uint4*>(&t)       = *reinterpret_cast<const uint4*>(r);
  *(reinterpret_cast<uint4*>(&t) + 1) = *reinterpret_cast<const uint4*>(r + 16);
  return __builtin_bit_cast(v16bf, t);
}

// B fragment (32x16 bf16) from a TRANSPOSED store Bt[N][K]:
// lane L holds col N=L&15; K-halves interleave like A: lanes 0-15 -> K {0..7,16..23}, lanes 16-31 -> K {8..15,24..31}.
static __device__ __forceinline__ v16bf load_bT(const unsigned short* baseT, int ldb, int lane) {
  const unsigned short* r = baseT + (lane & 15) * ldb + (lane >> 4) * 8;
  v16us t;
  *reinterpret_cast<uint4*>(&t)       = *reinterpret_cast<const uint4*>(r);
  *(reinterpret_cast<uint4*>(&t) + 1) = *reinterpret_cast<const uint4*>(r + 16);
  return __builtin_bit_cast(v16bf, t);
}

static __device__ __forceinline__ v8f wmma_bf16(v16bf a, v16bf b, v8f c) {
  return __builtin_amdgcn_wmma_f32_16x16x32_bf16(false, a, false, b, (short)0, c, false, false);
}

static __device__ __forceinline__ float wsum32(float v) {
  #pragma unroll
  for (int m = 16; m; m >>= 1) v += __shfl_xor(v, m, 32);
  return v;
}
static __device__ __forceinline__ float hmax16(float v) {
  #pragma unroll
  for (int m = 8; m; m >>= 1) v = fmaxf(v, __shfl_xor(v, m, 32));
  return v;
}
static __device__ __forceinline__ float hsum16(float v) {
  #pragma unroll
  for (int m = 8; m; m >>= 1) v += __shfl_xor(v, m, 32);
  return v;
}

// Tensor Data Mover: issue one descriptor (wave-uniform args). Arity differs by toolchain.
static __device__ __forceinline__ void tdm_load(v4u g0, v8i g1, v4i g2, v4i g3) {
#if defined(__clang_major__) && (__clang_major__ >= 23)
  v8i z8 = {0, 0, 0, 0, 0, 0, 0, 0};
  __builtin_amdgcn_tensor_load_to_lds(g0, g1, g2, g3, z8, 0);
#else
  __builtin_amdgcn_tensor_load_to_lds(g0, g1, g2, g3, 0);
#endif
}

// ======================================================================
// Kernel 1: LN1 + window attention + out-proj + residual -> y1 [P_TOK][96] f32
// One workgroup per window. 256 threads = 8 waves.
// ======================================================================
__global__ __launch_bounds__(256) void swin_attn_kernel(
    const float* __restrict__ x, const float* __restrict__ ln1_g, const float* __restrict__ ln1_b,
    const float* __restrict__ w_qkv, const float* __restrict__ w_out, const float* __restrict__ b_out,
    float* __restrict__ y1)
{
  extern __shared__ char smem[];
  unsigned short* sQ  = (unsigned short*)smem;               // [3][224][32] bf16
  unsigned short* sK  = sQ  + HEADS * T_PAD * HD;            // [3][224][32]
  unsigned short* sVT = sK  + HEADS * T_PAD * HD;            // [3][32][224] (V transposed)
  float* sRawX = (float*)smem;                               // TDM stage [96][216] f32 (aliases sQ/sK)
  char* regionR = (char*)(sVT + HEADS * T_PAD * HD);
  unsigned short* sXN    = (unsigned short*)regionR;         // [224][96]  (phase A)
  unsigned short* sWqkvT = sXN + T_PAD * C_DIM;              // [288][96]  (phase A)
  unsigned short* sP  = (unsigned short*)regionR;            // [8 waves][16][224] (phase B)
  unsigned short* sAO = sP + 8 * 16 * T_PAD;                 // [224][96]  (phase B)
  unsigned short* sWoutT = (unsigned short*)(regionR + REGION_BYTES);  // [96][96]
  float* sBout = (float*)(sWoutT + C_DIM * C_DIM);           // [96]

  const int tid  = threadIdx.x;
  const int lane = tid & 31;
  const int wave = __builtin_amdgcn_readfirstlane(tid >> 5);   // provably wave-uniform
  const int win  = blockIdx.x;
  const int dbase = (win >> 6) * 6, hbase = ((win >> 3) & 7) * 6, wbase = (win & 7) * 6;

  // ---- TDM: async-DMA the 4D window tile (6x6x6 spatial x 96 channels) into LDS ----
  if (wave == 0) {
    int p0 = (dbase * 48 + hbase) * 48 + wbase;
    unsigned long long ga = (unsigned long long)(const void*)x + (unsigned long long)p0 * 4ull;
    unsigned lds_off = (unsigned)(unsigned long long)(void*)sRawX;
    v4u g0 = { 1u, lds_off, (unsigned)ga,
               ((unsigned)(ga >> 32) & 0x1FFFFFFu) | (2u << 30) };   // addr[56:32] | type=2
    v8i g1 = { (int)(2u << 16),          // data_size = 4B
               (int)(48u << 16),         // tensor_dim0 = 48
               (int)(48u << 16),         // tensor_dim1 = 48
               (int)(6u  << 16),         // tile_dim0 = 6
               (int)(6u | (6u << 16)),   // tile_dim1 = 6, tile_dim2 = 6
               48,                       // tensor_dim0_stride
               (int)(2304u << 16),       // tensor_dim1_stride = 48*48
               0 };
    v4i g2 = { 48, 96, 110592, (int)(96u << 16) };  // dim2, dim3, dim2_stride, tile_dim3
    v4i g3 = { 0, 0, 0, 0 };
    tdm_load(g0, g1, g2, g3);            // LDS order: [c][t]
  }

  // ---- stage weights (transposed, bf16) while the TDM runs ----
  for (int i = tid; i < QKV_N * C_DIM; i += 256) {
    int n = i / C_DIM, k = i - n * C_DIM;
    sWqkvT[i] = f2bf(w_qkv[k * QKV_N + n]);
  }
  for (int i = tid; i < C_DIM * C_DIM; i += 256) {
    int n = i / C_DIM, k = i - n * C_DIM;
    sWoutT[i] = f2bf(w_out[k * C_DIM + n]);
  }
  if (tid < C_DIM) sBout[tid] = b_out[tid];

  if (wave == 0) __builtin_amdgcn_s_wait_tensorcnt(0);
  __syncthreads();

  // ---- LayerNorm1 from the LDS-staged tile ----
  for (int t = wave; t < T_PAD; t += 8) {
    const int c0 = lane * 3;
    if (t < T_WIN) {
      float xv[3];
      #pragma unroll
      for (int j = 0; j < 3; ++j) xv[j] = sRawX[(c0 + j) * T_WIN + t];
      float s  = wsum32(xv[0] + xv[1] + xv[2]);
      float s2 = wsum32(xv[0]*xv[0] + xv[1]*xv[1] + xv[2]*xv[2]);
      float mu  = s * (1.0f / 96.0f);
      float var = s2 * (1.0f / 96.0f) - mu * mu;
      float rstd = rsqrtf(var + 1e-5f);
      #pragma unroll
      for (int j = 0; j < 3; ++j) {
        int c = c0 + j;
        sXN[t * C_DIM + c] = f2bf((xv[j] - mu) * rstd * ln1_g[c] + ln1_b[c]);
      }
    } else {
      #pragma unroll
      for (int j = 0; j < 3; ++j) sXN[t * C_DIM + c0 + j] = 0;
    }
  }
  __syncthreads();   // sRawX dead; sQ/sK/sVT writable

  // ---- Q/K via transposed GEMM: (WqkvT[0:192] x XN^T) -> lane = token, rows = dims
  //      => one b128 store per lane into row-major Q/K.
  for (int task = wave; task < 12 * 14; task += 8) {
    int mi = task / 14, ni = task % 14;     // mi: channel tiles 0..11 (Q:0-5, K:6-11), ni: token tiles
    v8f acc = {};
    #pragma unroll
    for (int ks = 0; ks < 3; ++ks) {
      v16bf a = load_a (sWqkvT + mi * 16 * C_DIM + ks * 32, C_DIM, lane);
      v16bf b = load_bT(sXN    + ni * 16 * C_DIM + ks * 32, C_DIM, lane);
      acc = wmma_bf16(a, b, acc);
    }
    int hsel = (mi % 6) >> 1;                         // scalar
    int dd0  = ((mi & 1) << 4) + (lane >> 4) * 8;     // dim base within head
    int tok  = ni * 16 + (lane & 15);
    unsigned short* dst = (mi < 6 ? sQ : sK) + (hsel * T_PAD + tok) * HD + dd0;
    union { unsigned short vs[8]; uint4 q; } u;
    #pragma unroll
    for (int r = 0; r < 8; ++r) u.vs[r] = f2bf(acc[r]);
    *reinterpret_cast<uint4*>(dst) = u.q;
  }
  // ---- V via normal GEMM: lane = channel, rows = tokens -> b128 store into V^T
  for (int task = wave; task < 14 * 6; task += 8) {
    int mi = task / 6, ni = task % 6;       // mi: token tiles, ni: V channel tiles
    v8f acc = {};
    #pragma unroll
    for (int ks = 0; ks < 3; ++ks) {
      v16bf a = load_a (sXN    + mi * 16 * C_DIM + ks * 32, C_DIM, lane);
      v16bf b = load_bT(sWqkvT + (192 + ni * 16) * C_DIM + ks * 32, C_DIM, lane);
      acc = wmma_bf16(a, b, acc);
    }
    int hsel = ni >> 1;                               // scalar
    int dd   = ((ni & 1) << 4) + (lane & 15);
    int m0   = mi * 16 + (lane >> 4) * 8;
    union { unsigned short vs[8]; uint4 q; } u;
    #pragma unroll
    for (int r = 0; r < 8; ++r) u.vs[r] = f2bf(acc[r]);
    *reinterpret_cast<uint4*>(sVT + (hsel * HD + dd) * T_PAD + m0) = u.q;
  }
  __syncthreads();   // retires phase A of the union region

  // ---- attention: 42 tasks = (head, 16-row tile) ----
  for (int task = wave; task < HEADS * 14; task += 8) {
    int hh = task / 14, mi = task % 14;     // scalar
    v16bf aq = load_a(sQ + hh * T_PAD * HD + mi * 16 * HD, HD, lane);
    v8f st[14];
    #pragma unroll
    for (int kt = 0; kt < 14; ++kt) {
      v16bf b = load_bT(sK + hh * T_PAD * HD + kt * 16 * HD, HD, lane);
      v8f z = {};
      st[kt] = wmma_bf16(aq, b, z);
    }
    const int colb  = lane & 15;
    const int mloc0 = (lane >> 4) * 8;
    unsigned short* pslot = sP + wave * 16 * T_PAD;
    #pragma unroll
    for (int r = 0; r < 8; ++r) {
      float vals[14], mx = -1e30f;
      #pragma unroll
      for (int kt = 0; kt < 14; ++kt) {
        float v = st[kt][r] * SCALE_F;
        if (kt == 13 && colb >= 8) v = -1e30f;   // mask padded keys 216..223
        vals[kt] = v; mx = fmaxf(mx, v);
      }
      mx = hmax16(mx);
      float ssum = 0.f;
      #pragma unroll
      for (int kt = 0; kt < 14; ++kt) { float e = __expf(vals[kt] - mx); vals[kt] = e; ssum += e; }
      ssum = hsum16(ssum);
      float inv = 1.0f / ssum;
      int mloc = mloc0 + r;
      #pragma unroll
      for (int kt = 0; kt < 14; ++kt)
        pslot[mloc * T_PAD + kt * 16 + colb] = f2bf(vals[kt] * inv);
    }
    asm volatile("s_wait_dscnt 0" ::: "memory");   // own-slot store -> fragment reload
    // O^T = V^T x P^T: lane = q-token, rows = dims -> b128 store into sAO[token][ch]
    #pragma unroll
    for (int md = 0; md < 2; ++md) {
      v8f oc = {};
      #pragma unroll
      for (int ks = 0; ks < 7; ++ks) {
        v16bf a = load_a (sVT + (hh * HD + md * 16) * T_PAD + ks * 32, T_PAD, lane);
        v16bf b = load_bT(pslot + ks * 32, T_PAD, lane);
        oc = wmma_bf16(a, b, oc);
      }
      int tok = mi * 16 + (lane & 15);
      int d0  = md * 16 + (lane >> 4) * 8;
      union { unsigned short vs[8]; uint4 q; } u;
      #pragma unroll
      for (int r = 0; r < 8; ++r) u.vs[r] = f2bf(oc[r]);
      *reinterpret_cast<uint4*>(sAO + tok * C_DIM + hh * HD + d0) = u.q;
    }
  }
  __syncthreads();

  // ---- out projection (transposed): W_out^T x AO^T -> lane = token, rows = channels
  for (int task = wave; task < 6 * 14; task += 8) {
    int mi = task / 14, ni = task % 14;     // mi: out-channel tiles, ni: token tiles
    v8f acc = {};
    #pragma unroll
    for (int ks = 0; ks < 3; ++ks) {
      v16bf a = load_a (sWoutT + mi * 16 * C_DIM + ks * 32, C_DIM, lane);
      v16bf b = load_bT(sAO    + ni * 16 * C_DIM + ks * 32, C_DIM, lane);
      acc = wmma_bf16(a, b, acc);
    }
    int t = ni * 16 + (lane & 15);
    if (t < T_WIN) {
      int j0 = t / 36, j1 = (t / 6) % 6, j2 = t % 6;
      int p  = ((dbase + j0) * 48 + (hbase + j1)) * 48 + (wbase + j2);
      int n0 = mi * 16 + (lane >> 4) * 8;
      union { float vs[8]; v4f f4[2]; } u;
      #pragma unroll
      for (int r = 0; r < 8; ++r)
        u.vs[r] = acc[r] + sBout[n0 + r] + x[(n0 + r) * P_TOK + p];   // + pre-LN shortcut
      *reinterpret_cast<v4f*>(y1 + p * C_DIM + n0)     = u.f4[0];
      *reinterpret_cast<v4f*>(y1 + p * C_DIM + n0 + 4) = u.f4[1];
    }
  }
}

// ======================================================================
// Kernel 2: LN2 + MLP (96->384 GELU ->96) + residual -> out [96][P_TOK] f32
// 64 tokens per workgroup, 1728 workgroups.
// ======================================================================
__global__ __launch_bounds__(256) void swin_mlp_kernel(
    const float* __restrict__ y1, const float* __restrict__ ln2_g, const float* __restrict__ ln2_b,
    const float* __restrict__ w1, const float* __restrict__ b1,
    const float* __restrict__ w2, const float* __restrict__ b2,
    float* __restrict__ out)
{
  extern __shared__ char smem[];
  unsigned short* sW1T = (unsigned short*)smem;       // [384][96] w1 transposed
  unsigned short* sW2T = sW1T + HID * C_DIM;          // [96][384] w2 transposed
  unsigned short* sXN  = sW2T + HID * C_DIM;          // [64][96]
  unsigned short* sH   = sXN + 64 * C_DIM;            // [64][384]
  float* sB1 = (float*)(sH + 64 * HID);               // [384]
  float* sB2 = sB1 + HID;                             // [96]
  float* sY  = sB2 + C_DIM;                           // [64][96] TDM-staged y1 tile

  const int tid  = threadIdx.x;
  const int lane = tid & 31;
  const int wave = __builtin_amdgcn_readfirstlane(tid >> 5);
  const int base = blockIdx.x * 64;

  // ---- TDM: async-DMA this block's contiguous y1 tile (64*96 f32 = 24KB) into LDS ----
  if (wave == 0) {
    unsigned long long ga = (unsigned long long)(const void*)(y1 + (size_t)base * C_DIM);
    unsigned lds_off = (unsigned)(unsigned long long)(void*)sY;
    v4u g0 = { 1u, lds_off, (unsigned)ga,
               ((unsigned)(ga >> 32) & 0x1FFFFFFu) | (2u << 30) };
    v8i g1 = { (int)(2u << 16),            // data_size = 4B
               (int)(6144u << 16),         // tensor_dim0 = 6144
               0,
               (int)(6144u << 16),         // tile_dim0 = 6144 (1D tile)
               0, 6144, 0, 0 };
    v4i g2 = { 0, 0, 0, 0 };
    v4i g3 = { 0, 0, 0, 0 };
    tdm_load(g0, g1, g2, g3);
  }

  for (int i = tid; i < HID * C_DIM; i += 256) {
    int n = i / C_DIM, k = i - n * C_DIM;
    sW1T[i] = f2bf(w1[k * HID + n]);
  }
  for (int i = tid; i < C_DIM * HID; i += 256) {
    int n = i / HID, k = i - n * HID;
    sW2T[i] = f2bf(w2[k * C_DIM + n]);
  }
  for (int i = tid; i < HID; i += 256) sB1[i] = b1[i];
  if (tid < C_DIM) sB2[tid] = b2[tid];

  if (wave == 0) __builtin_amdgcn_s_wait_tensorcnt(0);
  __syncthreads();

  // ---- LayerNorm2 (reads the staged tile) ----
  for (int tl = wave; tl < 64; tl += 8) {
    int c0 = lane * 3;
    float xv[3];
    #pragma unroll
    for (int j = 0; j < 3; ++j) xv[j] = sY[tl * C_DIM + c0 + j];
    float s  = wsum32(xv[0] + xv[1] + xv[2]);
    float s2 = wsum32(xv[0]*xv[0] + xv[1]*xv[1] + xv[2]*xv[2]);
    float mu  = s * (1.0f / 96.0f);
    float var = s2 * (1.0f / 96.0f) - mu * mu;
    float rstd = rsqrtf(var + 1e-5f);
    #pragma unroll
    for (int j = 0; j < 3; ++j) {
      int c = c0 + j;
      sXN[tl * C_DIM + c] = f2bf((xv[j] - mu) * rstd * ln2_g[c] + ln2_b[c]);
    }
  }
  __syncthreads();

  // ---- GEMM1 (transposed): W1^T x XN^T + GELU -> lane = token, rows = hidden ch
  for (int task = wave; task < 24 * 4; task += 8) {
    int mi = task / 4, ni = task % 4;       // mi: hidden tiles, ni: token tiles
    v8f acc = {};
    #pragma unroll
    for (int ks = 0; ks < 3; ++ks) {
      v16bf a = load_a (sW1T + mi * 16 * C_DIM + ks * 32, C_DIM, lane);
      v16bf b = load_bT(sXN  + ni * 16 * C_DIM + ks * 32, C_DIM, lane);
      acc = wmma_bf16(a, b, acc);
    }
    int tok = ni * 16 + (lane & 15);
    int n0  = mi * 16 + (lane >> 4) * 8;
    union { unsigned short vs[8]; uint4 q; } u;
    #pragma unroll
    for (int r = 0; r < 8; ++r) {
      float h = acc[r] + sB1[n0 + r];
      u.vs[r] = f2bf(0.5f * h * (1.0f + erff(h * 0.70710678118654752f)));
    }
    *reinterpret_cast<uint4*>(sH + tok * HID + n0) = u.q;   // one b128 store
  }
  __syncthreads();

  // ---- GEMM2 [64,384]x[384,96] + bias + residual; contiguous-p channel-major store
  for (int task = wave; task < 4 * 6; task += 8) {
    int mi = task / 6, ni = task % 6;
    v8f acc = {};
    #pragma unroll
    for (int ks = 0; ks < 12; ++ks) {
      v16bf a = load_a (sH   + mi * 16 * HID + ks * 32, HID, lane);
      v16bf b = load_bT(sW2T + ni * 16 * HID + ks * 32, HID, lane);
      acc = wmma_bf16(a, b, acc);
    }
    int n  = ni * 16 + (lane & 15);
    int m0 = mi * 16 + (lane >> 4) * 8;
    float bb = sB2[n];
    union { float vs[8]; v4f f4[2]; } u;
    #pragma unroll
    for (int r = 0; r < 8; ++r)
      u.vs[r] = acc[r] + bb + sY[(m0 + r) * C_DIM + n];
    *reinterpret_cast<v4f*>(out + n * P_TOK + base + m0)     = u.f4[0];
    *reinterpret_cast<v4f*>(out + n * P_TOK + base + m0 + 4) = u.f4[1];
  }
}

// ======================================================================
extern "C" void kernel_launch(void* const* d_in, const int* in_sizes, int n_in,
                              void* d_out, int out_size, void* d_ws, size_t ws_size,
                              hipStream_t stream) {
  const float* x     = (const float*)d_in[0];
  const float* ln1_g = (const float*)d_in[1];
  const float* ln1_b = (const float*)d_in[2];
  const float* w_qkv = (const float*)d_in[3];
  const float* w_out = (const float*)d_in[4];
  const float* b_out = (const float*)d_in[5];
  const float* ln2_g = (const float*)d_in[6];
  const float* ln2_b = (const float*)d_in[7];
  const float* w1    = (const float*)d_in[8];
  const float* b1    = (const float*)d_in[9];
  const float* w2    = (const float*)d_in[10];
  const float* b2    = (const float*)d_in[11];

  float* y1 = (float*)d_ws;   // [110592][96] f32 attention+residual intermediate

  // LDS: qkv bf16 (129024 B, aliased by the 82944 B TDM x-stage) + union region (100352 B)
  //      + w_outT (18432 B) + b_out (384 B)  = 248192 B
  size_t lds1 = (size_t)(3 * HEADS * T_PAD * HD * 2) + REGION_BYTES
              + (size_t)(C_DIM * C_DIM * 2) + (size_t)(C_DIM * 4);
  swin_attn_kernel<<<512, 256, lds1, stream>>>(x, ln1_g, ln1_b, w_qkv, w_out, b_out, y1);

  // LDS: w1T + w2T (147456 B) + xn (12288 B) + h (49152 B) + biases (1920 B) + y1 tile (24576 B)
  size_t lds2 = (size_t)(2 * HID * C_DIM * 2) + (size_t)(64 * C_DIM * 2)
              + (size_t)(64 * HID * 2) + (size_t)((HID + C_DIM) * 4)
              + (size_t)(64 * C_DIM * 4);
  swin_mlp_kernel<<<P_TOK / 64, 256, lds2, stream>>>(y1, ln2_g, ln2_b, w1, b1, w2, b2,
                                                     (float*)d_out);
}
